// CRF_SelfAttention_65627100283470
// MI455X (gfx1250) — compile-verified, matrix operand order checked by reading, not run on hardware
//
#include <hip/hip_runtime.h>

// ---------------------------------------------------------------------------
// CRF multi-scale windowed self-attention forward for MI455X (gfx1250).
// All matrix math on v_wmma_f32_16x16x32_f16 (f16 in, f32 accumulate).
// Dead code (marginal / unary / compat / mp_*) eliminated: the final output
// depends only on context (softmax rows sum to 1 under the C-axis sum).
//
// Data-movement strategy (from the ISA 7.12.2 fragment layout):
//  - every lane's fragment = two contiguous 8-half runs -> 2x b128 loads
//  - weights pre-transposed so B-fragments are contiguous loads
//  - V projection stored transposed (one b128 store per lane) so the
//    attention P@V B-fragments are contiguous loads as well
// ---------------------------------------------------------------------------

typedef _Float16 f16;
typedef _Float16 v16h __attribute__((ext_vector_type(16)));
typedef _Float16 v8h  __attribute__((ext_vector_type(8)));
typedef float    v8f  __attribute__((ext_vector_type(8)));

#define F_     18
#define P_     128
#define H_     256
#define HEADS_ 8
#define HD_    32
#define NCLS_  625
#define TOK    (F_ * P_)   // 2304 tokens

// lane L: row/col = L%16 ; half-select hs = L/16
// element i -> k = i + 8*hs + (i>=8 ? 8 : 0)  == two contiguous runs of 8.
__device__ __forceinline__ v16h load_frag(const f16* __restrict__ base, int ld) {
  int lane = threadIdx.x & 31;
  int r = lane & 15, hs = lane >> 4;
  const f16* p = base + (size_t)r * ld + 8 * hs;
  v8h lo = *(const v8h*)(p);
  v8h hi = *(const v8h*)(p + 16);
  return __builtin_shufflevector(lo, hi, 0, 1, 2, 3, 4, 5, 6, 7,
                                 8, 9, 10, 11, 12, 13, 14, 15);
}

__device__ __forceinline__ v8f wmma_f16(v16h a, v16h b, v8f c) {
  return __builtin_amdgcn_wmma_f32_16x16x32_f16(false, a, false, b, (short)0, c,
                                                false, false);
}

// ---------------------------------------------------------------------------
// GEMM: C[M,N] = A[M,K](f16) @ B[K,N] + bias, with B given pre-transposed as
// BT[N,K] (f16, row-major).  One wave computes a 16x64 C strip: one A-frag
// feeds 4 WMMAs per 32-wide k-step.
// OUT: 0 = f32 row-major, 1 = f16 row-major, 2 = f16 transposed [N,M].
// ---------------------------------------------------------------------------
template <int OUT>
__global__ void k_gemm(const f16* __restrict__ A, const f16* __restrict__ BT,
                       const float* __restrict__ bias,
                       float* __restrict__ Cf, f16* __restrict__ Ch,
                       int M, int N, int K) {
  int wave = blockIdx.x * (blockDim.x >> 5) + (threadIdx.x >> 5);
  int ng = N >> 6;                       // groups of 64 columns
  if (wave >= (M >> 4) * ng) return;     // wave-uniform: EXEC stays all-ones
  int tm = wave / ng, tn4 = (wave % ng) * 4;

  const f16* a0 = A + (size_t)tm * 16 * K;
  v8f acc[4] = {};
  for (int k0 = 0; k0 < K; k0 += 32) {
    __builtin_prefetch(a0 + k0 + 64, 0, 1);      // global_prefetch_b8
    v16h a = load_frag(a0 + k0, K);
#pragma unroll
    for (int j = 0; j < 4; ++j) {
      v16h b = load_frag(BT + (size_t)(tn4 + j) * 16 * K + k0, K);
      acc[j] = wmma_f16(a, b, acc[j]);
    }
  }

  int lane = threadIdx.x & 31;
  int nl = lane & 15, hs = lane >> 4;
#pragma unroll
  for (int j = 0; j < 4; ++j) {
    int n = (tn4 + j) * 16 + nl;
    float bn = bias[n];
    if (OUT == 2) {                      // transposed f16: 8 contiguous m's
      v8h pack;
#pragma unroll
      for (int g = 0; g < 8; ++g) pack[g] = (f16)(acc[j][g] + bn);
      *(v8h*)(Ch + (size_t)n * M + tm * 16 + 8 * hs) = pack;
    } else {
#pragma unroll
      for (int g = 0; g < 8; ++g) {
        int m = tm * 16 + g + 8 * hs;
        float v = acc[j][g] + bn;
        if (OUT == 1) Ch[(size_t)m * N + n] = (f16)v;
        else          Cf[(size_t)m * N + n] = v;
      }
    }
  }
}

// ---------------------------------------------------------------------------
// Windowed MHSA core: one wave per (16-query tile, head, window).
// Q,K row-major [TOK,H]; V transposed [H,TOK] -> all fragments contiguous.
// ---------------------------------------------------------------------------
__global__ void k_attn(const f16* __restrict__ Q, const f16* __restrict__ Kd,
                       const f16* __restrict__ VT, f16* __restrict__ O, int S) {
  extern __shared__ float sc[];            // [16][S] scores / probs
  int qt = blockIdx.x, h = blockIdx.y, w = blockIdx.z;
  int lane = threadIdx.x & 31;
  int r = lane & 15, hs = lane >> 4;
  int row0 = w * P_;                       // window w covers tokens [w*P, w*P+S)

  v16h qf = load_frag(Q + (size_t)(row0 + qt * 16) * H_ + h * HD_, H_);
  const float scl = 0.17677669529663689f;  // 1/sqrt(hd=32)

  for (int kb = 0; kb < (S >> 4); ++kb) {
    // B^T of Q*K^T is the K tile itself -> same fragment pattern as A.
    v16h kf = load_frag(Kd + (size_t)(row0 + kb * 16) * H_ + h * HD_, H_);
    v8f s = {};
    s = wmma_f16(qf, kf, s);
#pragma unroll
    for (int g = 0; g < 8; ++g)
      sc[(g + 8 * hs) * S + kb * 16 + r] = s[g] * scl;
  }
  __syncthreads();

  if (lane < 16) {                         // one lane per query row
    float mx = -3.4e38f;
    for (int j = 0; j < S; ++j) mx = fmaxf(mx, sc[lane * S + j]);
    float sum = 0.0f;
    for (int j = 0; j < S; ++j) {
      float e = __expf(sc[lane * S + j] - mx);
      sc[lane * S + j] = e;
      sum += e;
    }
    float inv = 1.0f / sum;
    for (int j = 0; j < S; ++j) sc[lane * S + j] *= inv;
  }
  __syncthreads();                         // EXEC back to all-ones for WMMA

  v8f o0 = {}, o1 = {};
  for (int pb = 0; pb < (S >> 5); ++pb) {
    // P fragment from LDS: two contiguous 8-float runs per lane.
    const float* pr = &sc[r * S + pb * 32 + 8 * hs];
    v16h pf;
#pragma unroll
    for (int i = 0; i < 8; ++i) {
      pf[i]     = (f16)pr[i];
      pf[i + 8] = (f16)pr[16 + i];
    }
    // B[k,n] = V[row0+pb*32+k, h*32+j*16+n] = VT[(h*32+j*16+n)*TOK + ...]:
    // lane n reads contiguous k runs from VT row n.
    const f16* vb = VT + (size_t)(h * HD_) * TOK + row0 + pb * 32;
    v16h b0 = load_frag(vb, TOK);
    v16h b1 = load_frag(vb + (size_t)16 * TOK, TOK);
    o0 = wmma_f16(pf, b0, o0);
    o1 = wmma_f16(pf, b1, o1);
  }
  int n = lane & 15;
#pragma unroll
  for (int g = 0; g < 8; ++g) {
    int m = qt * 16 + g + 8 * hs;
    f16* orow = O + (size_t)(w * S + m) * H_ + h * HD_;
    orow[n]      = (f16)o0[g];
    orow[16 + n] = (f16)o1[g];
  }
}

// ---------------------------------------------------------------------------
// Elementwise / small kernels
// ---------------------------------------------------------------------------
__global__ void k_init(const float* __restrict__ emb, float* __restrict__ inp,
                       f16* __restrict__ inph, float* __restrict__ ctx) {
  int i = blockIdx.x * blockDim.x + threadIdx.x;
  if (i >= TOK * H_) return;
  float v = emb[i];
  inp[i] = v;
  inph[i] = (f16)v;
  ctx[i] = 0.0f;
}

__global__ void k_zero(float* __restrict__ b, int n) {
  int i = blockIdx.x * blockDim.x + threadIdx.x;
  if (i < n) b[i] = 0.0f;
}

// f32 [3,K,N] weights -> f16 transposed [3,N,K]
__global__ void k_cvt_t(const float* __restrict__ W, f16* __restrict__ WT,
                        int n3) {
  int i = blockIdx.x * blockDim.x + threadIdx.x;
  if (i >= n3) return;
  int s = i / (H_ * H_);
  int rem = i - s * (H_ * H_);
  int n = rem / H_, k = rem - n * H_;
  WT[i] = (f16)W[(size_t)s * H_ * H_ + (size_t)k * H_ + n];
}

// Overlap-add of per-window projection output into g with per-frame counts.
__global__ void k_overlap(float* __restrict__ g, const float* __restrict__ proj,
                          int scale, int nw) {
  int idx = blockIdx.x * blockDim.x + threadIdx.x;
  if (idx >= TOK * H_) return;
  int f = idx / (P_ * H_);
  int rem = idx % (P_ * H_);
  int w0 = f - scale + 1; if (w0 < 0) w0 = 0;
  int w1 = f;             if (w1 > nw - 1) w1 = nw - 1;
  float acc = 0.0f;
  for (int w = w0; w <= w1; ++w) {
    int j = f - w;  // frame index within window
    acc += proj[(size_t)(w * scale + j) * P_ * H_ + rem];
  }
  g[idx] += acc / (float)(w1 - w0 + 1);
}

// p[f] = mean_p sigmoid(inputs[f,p,:] . halt_W + halt_b)
__global__ void k_halt(const float* __restrict__ inp, const float* __restrict__ hw,
                       const float* __restrict__ hb, float* __restrict__ p) {
  __shared__ float red[128];
  int f = blockIdx.x, t = threadIdx.x;     // blockDim = 128 (= P_)
  const float* row = inp + (size_t)(f * P_ + t) * H_;
  float d = 0.0f;
  for (int k = 0; k < H_; ++k) d += row[k] * hw[k];
  d += hb[0];
  red[t] = 1.0f / (1.0f + __expf(-d));
  __syncthreads();
  for (int s = 64; s > 0; s >>= 1) {
    if (t < s) red[t] += red[t + s];
    __syncthreads();
  }
  if (t == 0) p[f] = red[0] / (float)P_;
}

// ACT-style halting bookkeeping; emits the context mixing weight wc[f].
__global__ void k_act(const float* __restrict__ p, float* __restrict__ ptn,
                      float* __restrict__ Rt, float* __restrict__ wc) {
  int f = threadIdx.x;
  if (f >= F_) return;
  float pt = ptn[f], rt = Rt[f], pf = p[f];
  float run  = (pt < 1.0f) ? 1.0f : 0.0f;
  float cand = pt + pf * run;
  float nh   = ((cand > 0.99f) ? 1.0f : 0.0f) * run;
  run        = ((cand <= 0.99f) ? 1.0f : 0.0f) * run;
  pt = pt + pf * run;
  rt = rt + nh * (1.0f - pt);
  pt = pt + nh * rt;
  ptn[f] = pt;
  Rt[f]  = rt;
  wc[f]  = pf * run + nh * rt;
}

// temporal = 0.25*g ; context = temporal*w + context*(1-w) ; inputs <- temporal
__global__ void k_ctx(const float* __restrict__ g, const float* __restrict__ wc,
                      float* __restrict__ ctx, float* __restrict__ inp,
                      f16* __restrict__ inph) {
  int idx = blockIdx.x * blockDim.x + threadIdx.x;
  if (idx >= TOK * H_) return;
  int f = idx / (P_ * H_);
  float t = 0.25f * g[idx];
  float w = wc[f];
  ctx[idx] = t * w + ctx[idx] * (1.0f - w);
  inp[idx] = t;
  inph[idx] = (f16)t;
}

// sv[h] = sum over all 2304 tokens of context[:,h]
__global__ void k_rowsum(const float* __restrict__ ctx, float* __restrict__ sv) {
  __shared__ float red[256];
  int h = blockIdx.x, t = threadIdx.x;
  float a = 0.0f;
  for (int r = t; r < TOK; r += 256) a += ctx[(size_t)r * H_ + h];
  red[t] = a;
  __syncthreads();
  for (int s = 128; s > 0; s >>= 1) {
    if (t < s) red[t] += red[t + s];
    __syncthreads();
  }
  if (t == 0) sv[h] = red[0];
}

__global__ void k_classes(const float* __restrict__ sv, const float* __restrict__ W,
                          const float* __restrict__ b, float* __restrict__ out) {
  int n = blockIdx.x * blockDim.x + threadIdx.x;
  if (n >= NCLS_) return;
  float a = b[n];
  for (int h = 0; h < H_; ++h) a += sv[h] * W[(size_t)h * NCLS_ + n];
  out[n] = a;
}

// ---------------------------------------------------------------------------
// Host orchestration
// ---------------------------------------------------------------------------
extern "C" void kernel_launch(void* const* d_in, const int* in_sizes, int n_in,
                              void* d_out, int out_size, void* d_ws, size_t ws_size,
                              hipStream_t stream) {
  (void)in_sizes; (void)n_in; (void)out_size; (void)ws_size;
  const float* emb    = (const float*)d_in[0];
  const float* halt_W = (const float*)d_in[3];
  const float* halt_b = (const float*)d_in[4];
  const float* cls_W  = (const float*)d_in[8];
  const float* cls_b  = (const float*)d_in[9];
  const float* Wq = (const float*)d_in[10];
  const float* bq = (const float*)d_in[11];
  const float* Wk = (const float*)d_in[12];
  const float* bk = (const float*)d_in[13];
  const float* Wv = (const float*)d_in[14];
  const float* bv = (const float*)d_in[15];
  const float* Wo = (const float*)d_in[16];
  const float* bo = (const float*)d_in[17];

  char* ws = (char*)d_ws;
  size_t off = 0;
  auto take = [&](size_t bytes) -> char* {
    char* p = ws + off;
    off += (bytes + 255) & ~(size_t)255;
    return p;
  };
  float* inp  = (float*)take((size_t)TOK * H_ * 4);
  f16*   inph = (f16*)  take((size_t)TOK * H_ * 2);
  f16*   qh   = (f16*)  take((size_t)TOK * H_ * 2);
  f16*   kh   = (f16*)  take((size_t)TOK * H_ * 2);
  f16*   vt   = (f16*)  take((size_t)TOK * H_ * 2);   // V transposed [H, TOK]
  f16*   attn = (f16*)  take((size_t)13 * 768 * H_ * 2);  // max scale-6 size
  float* proj = (float*)take((size_t)13 * 768 * H_ * 4);
  float* gbuf = (float*)take((size_t)TOK * H_ * 4);
  float* ctx  = (float*)take((size_t)TOK * H_ * 4);
  f16*   wqt  = (f16*)take((size_t)3 * H_ * H_ * 2);  // weights f16, transposed
  f16*   wkt  = (f16*)take((size_t)3 * H_ * H_ * 2);
  f16*   wvt  = (f16*)take((size_t)3 * H_ * H_ * 2);
  f16*   wot  = (f16*)take((size_t)3 * H_ * H_ * 2);
  float* pbuf = (float*)take(32 * 4);
  float* ptn  = (float*)take(32 * 4);
  float* Rt   = (float*)take(32 * 4);
  float* wc   = (float*)take(32 * 4);
  float* sv   = (float*)take((size_t)H_ * 4);

  const int NE  = TOK * H_;
  const int NW3 = 3 * H_ * H_;

  // --- init (fresh state every call: graph-replay safe) ---
  k_init<<<(NE + 255) / 256, 256, 0, stream>>>(emb, inp, inph, ctx);
  k_zero<<<1, 64, 0, stream>>>(ptn, 32);
  k_zero<<<1, 64, 0, stream>>>(Rt, 32);
  k_cvt_t<<<(NW3 + 255) / 256, 256, 0, stream>>>(Wq, wqt, NW3);
  k_cvt_t<<<(NW3 + 255) / 256, 256, 0, stream>>>(Wk, wkt, NW3);
  k_cvt_t<<<(NW3 + 255) / 256, 256, 0, stream>>>(Wv, wvt, NW3);
  k_cvt_t<<<(NW3 + 255) / 256, 256, 0, stream>>>(Wo, wot, NW3);

  const int scales[3] = {2, 4, 6};
  for (int it = 0; it < 2; ++it) {
    // halting probability + ACT state update (from current inputs)
    k_halt<<<F_, 128, 0, stream>>>(inp, halt_W, halt_b, pbuf);
    k_act<<<1, 32, 0, stream>>>(pbuf, ptn, Rt, wc);

    k_zero<<<(NE + 255) / 256, 256, 0, stream>>>(gbuf, NE);

    for (int i = 0; i < 3; ++i) {
      int s = scales[i], nw = F_ - s + 1, S = s * P_;
      size_t wo = (size_t)i * H_ * H_;
      // Q/K/V projections computed once over all tokens (window-independent)
      int wavesQ = (TOK / 16) * (H_ / 64);   // 16x64 strip per wave
      int blkQ = (wavesQ + 3) / 4;
      k_gemm<1><<<blkQ, 128, 0, stream>>>(inph, wqt + wo, bq + i * H_,
                                          nullptr, qh, TOK, H_, H_);
      k_gemm<1><<<blkQ, 128, 0, stream>>>(inph, wkt + wo, bk + i * H_,
                                          nullptr, kh, TOK, H_, H_);
      k_gemm<2><<<blkQ, 128, 0, stream>>>(inph, wvt + wo, bv + i * H_,
                                          nullptr, vt, TOK, H_, H_);
      // per-window attention
      k_attn<<<dim3(S / 16, HEADS_, nw), 32, (size_t)16 * S * 4, stream>>>(
          qh, kh, vt, attn, S);
      // per-window output projection
      int Mo = nw * S;
      int wavesO = (Mo / 16) * (H_ / 64);
      k_gemm<0><<<(wavesO + 3) / 4, 128, 0, stream>>>(
          attn, wot + wo, bo + i * H_, proj, nullptr, Mo, H_, H_);
      // overlap-add with per-frame counts
      k_overlap<<<(NE + 255) / 256, 256, 0, stream>>>(gbuf, proj, s, nw);
    }
    // temporal = 0.25*g ; context mix ; inputs <- temporal
    k_ctx<<<(NE + 255) / 256, 256, 0, stream>>>(gbuf, wc, ctx, inp, inph);
  }

  // classes = (sum_{p,f} context) @ cls_W + cls_b
  k_rowsum<<<H_, 256, 0, stream>>>(ctx, sv);
  k_classes<<<(NCLS_ + 255) / 256, 256, 0, stream>>>(sv, cls_W, cls_b,
                                                     (float*)d_out);
}
